// Attention_71339406787329
// MI455X (gfx1250) — compile-verified, matrix operand order checked by reading, not run on hardware
//
#include <hip/hip_runtime.h>
#include <hip/hip_bf16.h>
#include <cmath>

// ---------------------------------------------------------------------------
// MViT-style 3D rel-pos attention for MI455X (gfx1250, wave32, WMMA).
// All dense math runs through v_wmma_f32_16x16x32_bf16 (f32 accumulate).
// Attention is flash-style: the [48,1568,1568] score matrix is never
// materialized (per-head K+V = 392KB -> whole working set lives in L2).
// ---------------------------------------------------------------------------

typedef __bf16 bf16_t;
typedef __attribute__((ext_vector_type(16))) __bf16 v16bf;
typedef __attribute__((ext_vector_type(8)))  __bf16 v8bf;
typedef __attribute__((ext_vector_type(8)))  float  v8f;

#define NB    4
#define DD    8
#define HH    14
#define WW    14
#define CC    768
#define NH    12
#define HD    64
#define NSEQ  1568          // D*H*W
#define BHN   48            // B*NH
#define MM    6272          // B*NSEQ
#define N3C   2304

// Assemble a 16x32 bf16 WMMA fragment (A or B role) from a row-major row.
// CDNA5 layout: lane -> M (or N) = lane&15; half = lane>>4 selects K range.
// Per lane: 8 bf16 at k0+8*half and 8 bf16 at k0+16+8*half (both 16B runs).
__device__ inline v16bf load_frag(const bf16_t* __restrict__ row, int kk0, int half) {
    v8bf lo = *(const v8bf*)(row + kk0 + 8 * half);
    v8bf hi = *(const v8bf*)(row + kk0 + 16 + 8 * half);
    return __builtin_shufflevector(lo, hi, 0,1,2,3,4,5,6,7,8,9,10,11,12,13,14,15);
}

__device__ inline v8f wmma_bf16(v16bf a, v16bf b, v8f c) {
    return __builtin_amdgcn_wmma_f32_16x16x32_bf16(
        /*neg_a=*/false, a, /*neg_b=*/false, b,
        /*c_mod=*/(short)0, c, /*reuse_a=*/false, /*reuse_b=*/false);
}

// XOR-butterfly reductions over the 16-lane N-groups of the C/D layout.
// ds_swizzle group-of-32 mode: offset = xor<<10 | and_mask(0x1F).
__device__ inline float grp16_max(float v) {
    union { float f; int i; } u;
    u.f = v; u.i = __builtin_amdgcn_ds_swizzle(u.i, 0x041F); v = fmaxf(v, u.f);
    u.f = v; u.i = __builtin_amdgcn_ds_swizzle(u.i, 0x081F); v = fmaxf(v, u.f);
    u.f = v; u.i = __builtin_amdgcn_ds_swizzle(u.i, 0x101F); v = fmaxf(v, u.f);
    u.f = v; u.i = __builtin_amdgcn_ds_swizzle(u.i, 0x201F); v = fmaxf(v, u.f);
    return v;
}
__device__ inline float grp16_sum(float v) {
    union { float f; int i; } u;
    u.f = v; u.i = __builtin_amdgcn_ds_swizzle(u.i, 0x041F); v += u.f;
    u.f = v; u.i = __builtin_amdgcn_ds_swizzle(u.i, 0x081F); v += u.f;
    u.f = v; u.i = __builtin_amdgcn_ds_swizzle(u.i, 0x101F); v += u.f;
    u.f = v; u.i = __builtin_amdgcn_ds_swizzle(u.i, 0x201F); v += u.f;
    return v;
}

// ---------------------------------------------------------------------------
// Utility kernels
// ---------------------------------------------------------------------------
__global__ void k_f32_to_bf16(const float* __restrict__ in, bf16_t* __restrict__ out, int n) {
    int i = blockIdx.x * blockDim.x + threadIdx.x;
    if (i < n) out[i] = (bf16_t)in[i];
}

// out[n][k] = (bf16) in[k][n]   (in: [K][N] row-major)
__global__ void k_transpose_bf16(const float* __restrict__ in, bf16_t* __restrict__ out,
                                 int K, int N) {
    int idx = blockIdx.x * blockDim.x + threadIdx.x;
    if (idx >= K * N) return;
    int n = idx / K, k = idx % K;
    out[idx] = (bf16_t)in[(size_t)k * N + n];
}

// ---------------------------------------------------------------------------
// QKV GEMM: [6272,768] x [768,2304] (+bias). One wave per 16x64 output tile.
// Q stored raw bf16 [bh][n][hd]; K bf16 [bh][n][hd]; V transposed [bh][hd][n].
// ---------------------------------------------------------------------------
__global__ __launch_bounds__(32) void k_gemm_qkv(const bf16_t* __restrict__ X,
                                                 const bf16_t* __restrict__ Wt,
                                                 const float*  __restrict__ bias,
                                                 bf16_t* __restrict__ Qb,
                                                 bf16_t* __restrict__ Kb,
                                                 bf16_t* __restrict__ Vt) {
    const int mt = blockIdx.x;          // 392 row tiles
    const int n64 = blockIdx.y;         // 36 column tiles (64 wide)
    const int lane = threadIdx.x, l16 = lane & 15, half = lane >> 4;

    const bf16_t* Arow  = X  + (size_t)(mt * 16 + l16) * CC;
    const bf16_t* Brow0 = Wt + (size_t)(n64 * 64 + l16) * CC;

    v8f acc[4] = {};
    for (int kk = 0; kk < CC; kk += 32) {
        v16bf a = load_frag(Arow, kk, half);
#pragma unroll
        for (int ct = 0; ct < 4; ct++) {
            v16bf b = load_frag(Brow0 + (size_t)ct * 16 * CC, kk, half);
            acc[ct] = wmma_bf16(a, b, acc[ct]);
        }
    }
#pragma unroll
    for (int ct = 0; ct < 4; ct++) {
        int nG = n64 * 64 + ct * 16 + l16;
        int which = nG / CC;                 // uniform per block (64 | 768)
        int hc = nG - which * CC;
        int head = hc >> 6, c = hc & 63;
        float bv = bias[nG];
#pragma unroll
        for (int r = 0; r < 8; r++) {
            int mG = mt * 16 + r + 8 * half;
            int b_  = mG / NSEQ, ns = mG - b_ * NSEQ;   // tiles never straddle batches
            int bh = b_ * NH + head;
            float val = acc[ct][r] + bv;
            if (which == 0)       Qb[((size_t)bh * NSEQ + ns) * HD + c] = (bf16_t)val;
            else if (which == 1)  Kb[((size_t)bh * NSEQ + ns) * HD + c] = (bf16_t)val;
            else                  Vt[((size_t)bh * HD + c) * NSEQ + ns] = (bf16_t)val;
        }
    }
}

// ---------------------------------------------------------------------------
// Decomposed rel-pos bias: relAll[bh][qi][0:8]=d-term, [8:22]=h, [22:36]=w.
// Uses the *unscaled* q (matches reference: r_q = q, not q*scale).
// ---------------------------------------------------------------------------
__global__ void k_relbias(const bf16_t* __restrict__ Qb,
                          const float* __restrict__ rpd,
                          const float* __restrict__ rph,
                          const float* __restrict__ rpw,
                          float* __restrict__ relAll) {
    int blk = blockIdx.x;                 // bh*1568 + qi
    int bh = blk / NSEQ, qi = blk - bh * NSEQ;
    int d = qi / (HH * WW), rem = qi - d * (HH * WW), h = rem / WW, w = rem - h * WW;
    __shared__ float q[HD];
    int tid = threadIdx.x;
    if (tid < HD) q[tid] = (float)Qb[((size_t)bh * NSEQ + qi) * HD + tid];
    __syncthreads();
    if (tid < 36) {
        const float* tab; int row;
        if (tid < 8)       { tab = rpd; row = d - tid + (DD - 1); }
        else if (tid < 22) { tab = rph; row = h - (tid - 8) + (HH - 1); }
        else               { tab = rpw; row = w - (tid - 22) + (WW - 1); }
        const float* trow = tab + (size_t)row * HD;
        float s = 0.f;
#pragma unroll
        for (int c = 0; c < HD; c++) s += q[c] * trow[c];
        relAll[(size_t)blk * 36 + tid] = s;
    }
}

// ---------------------------------------------------------------------------
// Flash attention: one wave per (bh, 16-query tile). 49 iterations over
// 32-key tiles; 8 v_wmma per iteration (4 for S, 4 for P*V).
// ---------------------------------------------------------------------------
__global__ __launch_bounds__(32) void k_attention(const bf16_t* __restrict__ Qb,
                                                  const bf16_t* __restrict__ Kb,
                                                  const bf16_t* __restrict__ Vt,
                                                  const float*  __restrict__ relAll,
                                                  bf16_t* __restrict__ Ob) {
    const int qt = blockIdx.x;            // 98 query tiles
    const int bh = blockIdx.y;            // 48
    const int lane = threadIdx.x, l16 = lane & 15, half = lane >> 4;
    const float scale = 0.125f;           // hd^-0.5 (folded here; bias uses raw q)

    const bf16_t* Qrow = Qb + ((size_t)bh * NSEQ + qt * 16 + l16) * HD;
    v16bf qf0 = load_frag(Qrow, 0,  half);
    v16bf qf1 = load_frag(Qrow, 32, half);

    v8f acc[4] = {};
    float m_i[8], l_i[8];
#pragma unroll
    for (int r = 0; r < 8; r++) { m_i[r] = -__builtin_inff(); l_i[r] = 0.f; }

    __shared__ bf16_t Pl[16 * 72];        // 16x32 P tile, stride 72 -> conflict-free b128

    for (int kt = 0; kt < NSEQ / 32; kt++) {
        float s[2][8];
#pragma unroll
        for (int t = 0; t < 2; t++) {
            int key = kt * 32 + t * 16 + l16;
            const bf16_t* Krow = Kb + ((size_t)bh * NSEQ + key) * HD;
            v8f c = {};
            c = wmma_bf16(qf0, load_frag(Krow, 0,  half), c);
            c = wmma_bf16(qf1, load_frag(Krow, 32, half), c);
            int d2 = key / (HH * WW), krem = key - d2 * (HH * WW);
            int h2 = krem / WW, w2 = krem - h2 * WW;
#pragma unroll
            for (int r = 0; r < 8; r++) {
                int qi = qt * 16 + r + 8 * half;
                const float* rb = relAll + (size_t)(bh * NSEQ + qi) * 36;
                s[t][r] = c[r] * scale + rb[d2] + rb[8 + h2] + rb[22 + w2];
            }
        }
        if (kt + 1 < NSEQ / 32) {         // global_prefetch_b8 of next K/V tiles
            __builtin_prefetch(Kb + ((size_t)bh * NSEQ + (kt + 1) * 32 + lane) * HD, 0, 3);
            __builtin_prefetch(Vt + ((size_t)bh * HD + lane * 2) * NSEQ + (kt + 1) * 32, 0, 3);
        }
        // online softmax, P -> LDS in bf16 (A-matrix staging)
#pragma unroll
        for (int r = 0; r < 8; r++) {
            float mt2 = grp16_max(fmaxf(s[0][r], s[1][r]));
            float mnew = fmaxf(m_i[r], mt2);
            float alpha = __expf(m_i[r] - mnew);
            float p0 = __expf(s[0][r] - mnew);
            float p1 = __expf(s[1][r] - mnew);
            l_i[r] = l_i[r] * alpha + grp16_sum(p0 + p1);
            m_i[r] = mnew;
            acc[0][r] *= alpha; acc[1][r] *= alpha;
            acc[2][r] *= alpha; acc[3][r] *= alpha;
            int m = r + 8 * half;
            Pl[m * 72 + l16]      = (bf16_t)p0;
            Pl[m * 72 + 16 + l16] = (bf16_t)p1;
        }
        __syncthreads();
        v16bf pf = load_frag(&Pl[l16 * 72], 0, half);     // ds_load_b128 x2
#pragma unroll
        for (int ct = 0; ct < 4; ct++) {
            const bf16_t* Vrow = Vt + ((size_t)bh * HD + ct * 16 + l16) * NSEQ + kt * 32;
            acc[ct] = wmma_bf16(pf, load_frag(Vrow, 0, half), acc[ct]);
        }
        __syncthreads();
    }
    // epilogue: normalize, merge heads -> Ob[b][n][C] (bf16 for proj GEMM)
    int b_ = bh / NH, head = bh - b_ * NH;
#pragma unroll
    for (int r = 0; r < 8; r++) {
        float inv = 1.0f / l_i[r];
        int qi = qt * 16 + r + 8 * half;
        size_t base = ((size_t)(b_ * NSEQ + qi)) * CC + head * HD;
#pragma unroll
        for (int ct = 0; ct < 4; ct++)
            Ob[base + ct * 16 + l16] = (bf16_t)(acc[ct][r] * inv);
    }
}

// ---------------------------------------------------------------------------
// Output projection: [6272,768] x [768,768] + bias -> f32 d_out.
// ---------------------------------------------------------------------------
__global__ __launch_bounds__(32) void k_gemm_proj(const bf16_t* __restrict__ A,
                                                  const bf16_t* __restrict__ Wt,
                                                  const float*  __restrict__ bias,
                                                  float* __restrict__ out) {
    const int mt = blockIdx.x;            // 392
    const int n64 = blockIdx.y;           // 12
    const int lane = threadIdx.x, l16 = lane & 15, half = lane >> 4;

    const bf16_t* Arow  = A  + (size_t)(mt * 16 + l16) * CC;
    const bf16_t* Brow0 = Wt + (size_t)(n64 * 64 + l16) * CC;

    v8f acc[4] = {};
    for (int kk = 0; kk < CC; kk += 32) {
        v16bf a = load_frag(Arow, kk, half);
#pragma unroll
        for (int ct = 0; ct < 4; ct++) {
            v16bf b = load_frag(Brow0 + (size_t)ct * 16 * CC, kk, half);
            acc[ct] = wmma_bf16(a, b, acc[ct]);
        }
    }
#pragma unroll
    for (int ct = 0; ct < 4; ct++) {
        int nG = n64 * 64 + ct * 16 + l16;
        float bv = bias[nG];
#pragma unroll
        for (int r = 0; r < 8; r++) {
            int mG = mt * 16 + r + 8 * half;
            out[(size_t)mG * CC + nG] = acc[ct][r] + bv;
        }
    }
}

// ---------------------------------------------------------------------------
extern "C" void kernel_launch(void* const* d_in, const int* in_sizes, int n_in,
                              void* d_out, int out_size, void* d_ws, size_t ws_size,
                              hipStream_t stream) {
    const float* x      = (const float*)d_in[0];
    const float* qkv_w  = (const float*)d_in[1];
    const float* qkv_b  = (const float*)d_in[2];
    const float* proj_w = (const float*)d_in[3];
    const float* proj_b = (const float*)d_in[4];
    const float* rpd    = (const float*)d_in[5];
    const float* rph    = (const float*)d_in[6];
    const float* rpw    = (const float*)d_in[7];

    char* ws = (char*)d_ws;
    size_t off = 0;
    auto take = [&](size_t bytes) -> void* {
        void* p = ws + off;
        off += (bytes + 255) & ~(size_t)255;
        return p;
    };
    bf16_t* Xbf    = (bf16_t*)take((size_t)MM * CC * 2);
    bf16_t* Wqkvt  = (bf16_t*)take((size_t)N3C * CC * 2);
    bf16_t* Wprojt = (bf16_t*)take((size_t)CC * CC * 2);
    bf16_t* Qb     = (bf16_t*)take((size_t)BHN * NSEQ * HD * 2);
    bf16_t* Kb     = (bf16_t*)take((size_t)BHN * NSEQ * HD * 2);
    bf16_t* Vt     = (bf16_t*)take((size_t)BHN * NSEQ * HD * 2);
    float*  relAll = (float*) take((size_t)BHN * NSEQ * 36 * 4);
    bf16_t* Ob     = (bf16_t*)take((size_t)MM * CC * 2);

    int nx = MM * CC;
    k_f32_to_bf16<<<(nx + 255) / 256, 256, 0, stream>>>(x, Xbf, nx);
    k_transpose_bf16<<<(CC * N3C + 255) / 256, 256, 0, stream>>>(qkv_w, Wqkvt, CC, N3C);
    k_transpose_bf16<<<(CC * CC + 255) / 256, 256, 0, stream>>>(proj_w, Wprojt, CC, CC);

    k_gemm_qkv<<<dim3(MM / 16, N3C / 64), 32, 0, stream>>>(Xbf, Wqkvt, qkv_b, Qb, Kb, Vt);
    k_relbias<<<BHN * NSEQ, 64, 0, stream>>>(Qb, rpd, rph, rpw, relAll);
    k_attention<<<dim3(NSEQ / 16, BHN), 32, 0, stream>>>(Qb, Kb, Vt, relAll, Ob);
    k_gemm_proj<<<dim3(MM / 16, CC / 64), 32, 0, stream>>>(Ob, Wprojt, proj_b, (float*)d_out);
}
